// GRU_RNN_73151882986172
// MI455X (gfx1250) — matrix operand order repaired and checked by reading, not compiled, not run on hardware
//
#include <hip/hip_runtime.h>
#include <hip/hip_bf16.h>

typedef __attribute__((ext_vector_type(16))) __bf16 v16bf;
typedef __attribute__((ext_vector_type(8)))  float  v8f;

#define T_SEQ 1024
#define DIM   300
#define HID   512
#define VGV   40000
#define VSV   25000

// ---------- helpers ----------
__device__ __forceinline__ unsigned short f2bf(float f) {
    unsigned int u = __float_as_uint(f);
    unsigned int r = 0x7FFFu + ((u >> 16) & 1u);
    return (unsigned short)((u + r) >> 16);
}
__device__ __forceinline__ float bf2f(unsigned int us) {
    return __uint_as_float(us << 16);
}
__device__ __forceinline__ float sigmoidf(float x) {
    return 1.0f / (1.0f + __expf(-x));
}

union BFrag { unsigned int u[8]; v16bf v; };

// ---------- prep kernels ----------
__global__ void k_prep_x(const int* __restrict__ tokens, const float* __restrict__ X,
                         unsigned short* __restrict__ xb) {
    int e = blockIdx.x * 256 + threadIdx.x;            // T_SEQ*320 elements
    int t = e / 320, k = e % 320;
    float v = (k < DIM) ? X[(size_t)tokens[t] * DIM + k] : 0.0f;
    xb[e] = f2bf(v);
}

__global__ void k_prep_w1(const float* __restrict__ Wz1, const float* __restrict__ Wr1,
                          const float* __restrict__ W1, unsigned short* __restrict__ w1b) {
    int e = blockIdx.x * 256 + threadIdx.x;            // 320*1536 elements, layout [k][n]
    int k = e / 1536, n = e % 1536;
    int g = n / HID, j = n % HID;
    float v = 0.0f;
    if (k < DIM) {
        const float* W = (g == 0) ? Wz1 : ((g == 1) ? Wr1 : W1);
        v = W[(size_t)k * HID + j];
    }
    w1b[e] = f2bf(v);
}

__global__ void k_prep_ru(const float* Uz1, const float* Ur1, const float* U1,
                          const float* Wz2, const float* Uz2, const float* Wr2,
                          const float* Ur2, const float* W2, const float* U2,
                          unsigned short* __restrict__ rub) {
    size_t e = (size_t)blockIdx.x * 256 + threadIdx.x; // 2359296 elements
    const size_t S = (size_t)HID * HID;
    float v;
    if (e < 3 * S) {
        int seg = (int)(e / S); size_t o = e % S;
        const float* M = (seg == 0) ? Uz1 : ((seg == 1) ? Ur1 : U1);
        v = M[o];
    } else {
        size_t e2 = e - 3 * S;
        int seg = (int)(e2 / (2 * S)); size_t o = e2 % (2 * S);
        size_t k = o / HID, j = o % HID;
        const float* top = (seg == 0) ? Wz2 : ((seg == 1) ? Wr2 : W2);
        const float* bot = (seg == 0) ? Uz2 : ((seg == 1) ? Ur2 : U2);
        v = (k < HID) ? top[k * HID + j] : bot[(k - HID) * HID + j];
    }
    rub[e] = f2bf(v);
}

// ---------- WMMA GEMM: C[M][Nvalid] = A(bf16)[M][K] * B[K][N] (+bias) ----------
// Tile: BM=128 (8 waves x 16 rows), BN=128 (8 N-tiles), BK=32.
// A tile: async-copied global->LDS (double buffered, ASYNCcnt).
// B tile: f32 (or bf16) wide loads into regs, converted to bf16, transposed in LDS.
__global__ void __launch_bounds__(256) k_gemm(
        const unsigned short* __restrict__ Ab, int K, int lda,
        const void* __restrict__ Bsrc, int bIsF32, int ldb,
        const float* __restrict__ bias, float* __restrict__ Cout, int Nvalid) {
    // A: row stride 40 ushorts (80B, 16B-aligned for b128 async stores)
    __shared__ __align__(16) unsigned short As[2][128 * 40];
    __shared__ unsigned short Bs[128 * 34];   // transposed [n][k], k-pairs contiguous
    int tid  = threadIdx.x;
    int lane = tid & 31, wv = tid >> 5;
    int mBase = blockIdx.y * 128;
    int nb    = blockIdx.x * 128;
    const float*          Bf = (const float*)Bsrc;
    const unsigned short* Bb = (const unsigned short*)Bsrc;
    int hl = lane >> 4, ln = lane & 15;
    int nIter = K >> 5;

    v8f zero;
#pragma unroll
    for (int i = 0; i < 8; ++i) zero[i] = 0.0f;
    v8f acc[8];
#pragma unroll
    for (int i = 0; i < 8; ++i) acc[i] = zero;

    float4 bf4[4];
    uint4  bu4[2];

    // --- async A tile: 128x32 bf16 = 512 chunks of 16B, 2 per thread ---
    auto issueA = [&](int it, int buf) {
        int k0 = it << 5;
#pragma unroll
        for (int i = 0; i < 2; ++i) {
            int e = i * 256 + tid;
            int row = e >> 2, q = e & 3;
            unsigned ldsOff = (unsigned)(unsigned long long)(&As[buf][0]) + row * 80 + q * 16;
            const unsigned short* g = Ab + (size_t)(mBase + row) * lda + k0 + q * 8;
            asm volatile("global_load_async_to_lds_b128 %0, %1, off"
                         :: "v"(ldsOff), "v"(g) : "memory");
        }
    };
    // --- B tile loads into registers (wide) ---
    auto loadB = [&](int it) {
        int k0 = it << 5;
        if (bIsF32) {
#pragma unroll
            for (int i = 0; i < 4; ++i) {
                int e = i * 256 + tid;
                int k = e >> 5, c4 = e & 31;
                int col = nb + (c4 << 2);
                float4 t = {0.0f, 0.0f, 0.0f, 0.0f};
                if (col + 3 < Nvalid) {
                    t = *(const float4*)(Bf + (size_t)(k0 + k) * ldb + col);
                } else {
                    if (col + 0 < Nvalid) t.x = Bf[(size_t)(k0 + k) * ldb + col + 0];
                    if (col + 1 < Nvalid) t.y = Bf[(size_t)(k0 + k) * ldb + col + 1];
                    if (col + 2 < Nvalid) t.z = Bf[(size_t)(k0 + k) * ldb + col + 2];
                }
                bf4[i] = t;
            }
        } else {
#pragma unroll
            for (int i = 0; i < 2; ++i) {
                int e = i * 256 + tid;
                int k = e >> 4, c8 = e & 15;
                int col = nb + (c8 << 3);
                bu4[i] = *(const uint4*)(Bb + (size_t)(k0 + k) * ldb + col);
            }
        }
    };
    auto storeB = [&]() {
        if (bIsF32) {
#pragma unroll
            for (int i = 0; i < 4; ++i) {
                int e = i * 256 + tid;
                int k = e >> 5, c4 = e & 31;
                int n = c4 << 2;
                Bs[(n + 0) * 34 + k] = f2bf(bf4[i].x);
                Bs[(n + 1) * 34 + k] = f2bf(bf4[i].y);
                Bs[(n + 2) * 34 + k] = f2bf(bf4[i].z);
                Bs[(n + 3) * 34 + k] = f2bf(bf4[i].w);
            }
        } else {
#pragma unroll
            for (int i = 0; i < 2; ++i) {
                int e = i * 256 + tid;
                int k = e >> 4, c8 = e & 15;
                int n = c8 << 3;
                Bs[(n + 0) * 34 + k] = (unsigned short)(bu4[i].x & 0xFFFFu);
                Bs[(n + 1) * 34 + k] = (unsigned short)(bu4[i].x >> 16);
                Bs[(n + 2) * 34 + k] = (unsigned short)(bu4[i].y & 0xFFFFu);
                Bs[(n + 3) * 34 + k] = (unsigned short)(bu4[i].y >> 16);
                Bs[(n + 4) * 34 + k] = (unsigned short)(bu4[i].z & 0xFFFFu);
                Bs[(n + 5) * 34 + k] = (unsigned short)(bu4[i].z >> 16);
                Bs[(n + 6) * 34 + k] = (unsigned short)(bu4[i].w & 0xFFFFu);
                Bs[(n + 7) * 34 + k] = (unsigned short)(bu4[i].w >> 16);
            }
        }
    };

    issueA(0, 0);
    loadB(0);
    for (int it = 0; it < nIter; ++it) {
        storeB();
        if (it + 1 < nIter) {
            issueA(it + 1, (it + 1) & 1);              // overlap next A with compute
            asm volatile("s_wait_asynccnt 0x2" ::: "memory");  // tile 'it' done, next in flight
        } else {
            asm volatile("s_wait_asynccnt 0x0" ::: "memory");
        }
        __syncthreads();
        if (it + 1 < nIter) loadB(it + 1);             // next B overlaps WMMA

        const unsigned short* Asb = As[it & 1];
        BFrag a;
        int m = wv * 16 + ln;
#pragma unroll
        for (int j = 0; j < 8; ++j) {
            int kk = (j >> 2) * 16 + hl * 8 + ((j & 3) << 1);
            a.u[j] = *(const unsigned int*)&Asb[m * 40 + kk];
        }
#pragma unroll
        for (int nt = 0; nt < 8; ++nt) {
            BFrag b;
            int n = nt * 16 + ln;
#pragma unroll
            for (int j = 0; j < 8; ++j) {
                b.u[j] = *(const unsigned int*)&Bs[n * 34 + (j >> 2) * 16 + hl * 8 + ((j & 3) << 1)];
            }
            acc[nt] = __builtin_amdgcn_wmma_f32_16x16x32_bf16(
                false, a.v, false, b.v, (short)0, acc[nt], false, false);
        }
        __syncthreads();
    }

    // C layout: VGPR r -> M = r + 8*(lane/16), N = lane%16
#pragma unroll
    for (int nt = 0; nt < 8; ++nt) {
        int col = nb + nt * 16 + ln;
        if (col < Nvalid) {
            float bv = bias ? bias[col] : 0.0f;
#pragma unroll
            for (int r = 0; r < 8; ++r) {
                int row = mBase + wv * 16 + hl * 8 + r;
                Cout[(size_t)row * Nvalid + col] = acc[nt][r] + bv;
            }
        }
    }
}

// ---------- recurrent scan (batch-1, sequential) ----------
// matvec out[j] = sum_k vec[k]*W[k][j]: 64 groups of 8 cols (b128 weight loads)
// x 16-way K split; LDS partial reduce. 1024 threads.
__device__ __forceinline__ void mv(const unsigned short* __restrict__ Wb,
                                   const float* __restrict__ vec, int K,
                                   float* part, float* out) {
    int tid = threadIdx.x;
    int g = tid & 63, s = tid >> 6;
    int j0 = g << 3;
    int kseg = K >> 4;
    float a[8] = {0.f, 0.f, 0.f, 0.f, 0.f, 0.f, 0.f, 0.f};
    int kend = (s + 1) * kseg;
    for (int k = s * kseg; k < kend; ++k) {
        uint4 w = *(const uint4*)(Wb + (size_t)k * HID + j0);
        float v = vec[k];
        a[0] = fmaf(v, bf2f(w.x & 0xFFFFu), a[0]);
        a[1] = fmaf(v, bf2f(w.x >> 16),     a[1]);
        a[2] = fmaf(v, bf2f(w.y & 0xFFFFu), a[2]);
        a[3] = fmaf(v, bf2f(w.y >> 16),     a[3]);
        a[4] = fmaf(v, bf2f(w.z & 0xFFFFu), a[4]);
        a[5] = fmaf(v, bf2f(w.z >> 16),     a[5]);
        a[6] = fmaf(v, bf2f(w.w & 0xFFFFu), a[6]);
        a[7] = fmaf(v, bf2f(w.w >> 16),     a[7]);
    }
#pragma unroll
    for (int i = 0; i < 8; ++i) part[s * 512 + j0 + i] = a[i];
    __syncthreads();
    if (tid < 512) {
        float r = 0.0f;
#pragma unroll
        for (int i = 0; i < 16; ++i) r += part[i * 512 + tid];
        out[tid] = r;
    }
    __syncthreads();
}

__global__ void __launch_bounds__(1024) k_scan(const float* __restrict__ A1,
                                               const unsigned short* __restrict__ rub,
                                               unsigned short* __restrict__ h2sb) {
    __shared__ float h1[512], h2[512], zb[512], tmp[512], hcat[1024], pre[512];
    __shared__ float part[16 * 512];
    const size_t S = (size_t)HID * HID;
    const unsigned short* Uz1  = rub;
    const unsigned short* Ur1  = rub + S;
    const unsigned short* Uc1  = rub + 2 * S;
    const unsigned short* WUz2 = rub + 3 * S;
    const unsigned short* WUr2 = rub + 5 * S;
    const unsigned short* WUc2 = rub + 7 * S;
    int tid = threadIdx.x;
    if (tid < 512) { h1[tid] = 0.0f; h2[tid] = 0.0f; }
    __syncthreads();

    for (int t = 0; t < T_SEQ; ++t) {
        const float* a1t = A1 + (size_t)t * 1536;
        mv(Uz1, h1, HID, part, pre);
        if (tid < 512) zb[tid] = sigmoidf(a1t[tid] + pre[tid]);
        mv(Ur1, h1, HID, part, pre);
        if (tid < 512) {
            float r = sigmoidf(a1t[512 + tid] + pre[tid]);
            tmp[tid] = r * h1[tid];
        }
        __syncthreads();
        mv(Uc1, tmp, HID, part, pre);
        if (tid < 512) {
            float ht = tanhf(a1t[1024 + tid] + pre[tid]);
            float z = zb[tid];
            float h1n = z * ht + (1.0f - z) * h1[tid];
            hcat[tid] = h1n;
            hcat[512 + tid] = h2[tid];
        }
        __syncthreads();
        mv(WUz2, hcat, 1024, part, pre);
        if (tid < 512) zb[tid] = sigmoidf(pre[tid]);
        mv(WUr2, hcat, 1024, part, pre);
        if (tid < 512) {
            float r2 = sigmoidf(pre[tid]);
            hcat[512 + tid] = r2 * h2[tid];
        }
        __syncthreads();
        mv(WUc2, hcat, 1024, part, pre);
        if (tid < 512) {
            float ht2 = tanhf(pre[tid]);
            float z2 = zb[tid];
            float h2n = z2 * ht2 + (1.0f - z2) * h2[tid];
            h2[tid] = h2n;
            h1[tid] = hcat[tid];
            h2sb[(size_t)t * HID + tid] = f2bf(h2n);
        }
        __syncthreads();
    }
}

// ---------- per-row log-softmax, in place ----------
__global__ void __launch_bounds__(256) k_lsm(float* __restrict__ logits, int V) {
    __shared__ float red[256];
    int tid = threadIdx.x;
    float* p = logits + (size_t)blockIdx.x * V;
    float m = -3.4e38f;
    for (int i = tid; i < V; i += 256) m = fmaxf(m, p[i]);
    red[tid] = m; __syncthreads();
    for (int s = 128; s > 0; s >>= 1) {
        if (tid < s) red[tid] = fmaxf(red[tid], red[tid + s]);
        __syncthreads();
    }
    m = red[0]; __syncthreads();
    float sum = 0.0f;
    for (int i = tid; i < V; i += 256) sum += __expf(p[i] - m);
    red[tid] = sum; __syncthreads();
    for (int s = 128; s > 0; s >>= 1) {
        if (tid < s) red[tid] += red[tid + s];
        __syncthreads();
    }
    float lse = m + __logf(red[0]);
    for (int i = tid; i < V; i += 256) p[i] -= lse;
}

// ---------- launch ----------
extern "C" void kernel_launch(void* const* d_in, const int* in_sizes, int n_in,
                              void* d_out, int out_size, void* d_ws, size_t ws_size,
                              hipStream_t stream) {
    const int*   tokens = (const int*)d_in[0];
    const float* X   = (const float*)d_in[1];
    const float* Wz1 = (const float*)d_in[2];  const float* Uz1 = (const float*)d_in[3];
    const float* Wr1 = (const float*)d_in[4];  const float* Ur1 = (const float*)d_in[5];
    const float* W1  = (const float*)d_in[6];  const float* U1  = (const float*)d_in[7];
    const float* Wz2 = (const float*)d_in[8];  const float* Uz2 = (const float*)d_in[9];
    const float* Wr2 = (const float*)d_in[10]; const float* Ur2 = (const float*)d_in[11];
    const float* W2  = (const float*)d_in[12]; const float* U2  = (const float*)d_in[13];
    const float* Wg  = (const float*)d_in[14]; const float* bg  = (const float*)d_in[15];
    const float* Wsn = (const float*)d_in[16]; const float* bsn = (const float*)d_in[17];

    char* ws = (char*)d_ws;
    unsigned short* xb   = (unsigned short*)(ws);                 // 1024*320 bf16
    unsigned short* w1b  = (unsigned short*)(ws + 655360);        // 320*1536 bf16
    float*          A1   = (float*)(ws + 1638400);                // 1024*1536 f32
    unsigned short* rub  = (unsigned short*)(ws + 7929856);       // packed recurrent bf16
    unsigned short* h2sb = (unsigned short*)(ws + 12648448);      // 1024*512 bf16

    float* outG = (float*)d_out;
    float* outS = outG + (size_t)T_SEQ * VGV;

    k_prep_x <<<(T_SEQ * 320) / 256, 256, 0, stream>>>(tokens, X, xb);
    k_prep_w1<<<(320 * 1536) / 256, 256, 0, stream>>>(Wz1, Wr1, W1, w1b);
    k_prep_ru<<<2359296 / 256, 256, 0, stream>>>(Uz1, Ur1, U1, Wz2, Uz2, Wr2, Ur2, W2, U2, rub);

    // A1 = x_emb(bf16) @ [Wz1|Wr1|W1](bf16): M=1024, K=320, N=1536
    k_gemm<<<dim3(1536 / 128, T_SEQ / 128), 256, 0, stream>>>(
        xb, 320, 320, w1b, /*bIsF32=*/0, 1536, nullptr, A1, 1536);

    k_scan<<<1, 1024, 0, stream>>>(A1, rub, h2sb);

    // vocab logits: h2s(bf16)[1024,512] @ W(f32->bf16)[512,V] + bias
    k_gemm<<<dim3((VGV + 127) / 128, T_SEQ / 128), 256, 0, stream>>>(
        h2sb, HID, HID, Wg, /*bIsF32=*/1, VGV, bg, outG, VGV);
    k_gemm<<<dim3((VSV + 127) / 128, T_SEQ / 128), 256, 0, stream>>>(
        h2sb, HID, HID, Wsn, /*bIsF32=*/1, VSV, bsn, outS, VSV);

    k_lsm<<<T_SEQ, 256, 0, stream>>>(outG, VGV);
    k_lsm<<<T_SEQ, 256, 0, stream>>>(outS, VSV);
}
